// BC6SurrogateBlockLevel_44281112822321
// MI455X (gfx1250) — matrix operand order repaired and checked by reading, not compiled, not run on hardware
//
#include <hip/hip_runtime.h>

// ---------------------------------------------------------------------------
// BC6H soft-decode surrogate for MI455X (gfx1250, wave32).
//
// One wave handles 16 consecutive 4x4 blocks:
//   * softmax(32 logits) per block  -> rows of A (16x32 f16)
//   * partition bank (32x16)        -> B (f16)
//   * v_wmma_f32_16x16x32_f16       -> 16x16 soft-mask tile (f32 acc)
//   * per-lane decode (sigmoid / LUT lerp / lerp-blend / uf16 surrogate)
//   * LDS-staged transpose -> coalesced 256B global_store_b64 segments
// ---------------------------------------------------------------------------

typedef __attribute__((ext_vector_type(16))) _Float16 v16h;
typedef __attribute__((ext_vector_type(8)))  float    v8f;

#define IMG_H 4096
#define IMG_W 4096
#define BLK_X 1024          // blocks per image row
#define NB    1048576

// per-wave LDS layout (in floats)
#define LUT_OFF 0                  // 8 floats: weight LUT copy
#define EU_OFF  8                  // 192 floats: e_u[(M*3+ch)*4 + ep]
#define Y_OFF   200                // 48*18 floats: y[(c*16+M)*18 + t] (pad 18)
#define WAVE_STRIDE 1072           // 4288 B per wave, 16B aligned
#define WAVES_PER_WG 8

__device__ __forceinline__ float fast_sigmoid(float x) {
    return 1.0f / (1.0f + __expf(-x));
}

__global__ __launch_bounds__(256) void bc6_soft_decode(
    const float* __restrict__ endpoints,   // (NB,4,3)
    const float* __restrict__ indices,     // (NB,16)
    const float* __restrict__ logits,      // (NB,32)
    const float* __restrict__ bank,        // (32,16)
    const float* __restrict__ lut,         // (8)
    float* __restrict__ out)               // (3,4096,4096)
{
    __shared__ float smem[WAVES_PER_WG * WAVE_STRIDE];

    const int lane = threadIdx.x & 31;
    const int wave = threadIdx.x >> 5;
    float* ws = smem + wave * WAVE_STRIDE;

    const int blockBase = (blockIdx.x * WAVES_PER_WG + wave) * 16;

    // ---- per-wave LUT copy -------------------------------------------------
    if (lane < 8) ws[LUT_OFF + lane] = lut[lane];

    // ---- stage e_u into LDS (192 values, 6 per lane) -----------------------
    for (int f = lane; f < 192; f += 32) {
        int M   = f / 12;
        int rem = f - M * 12;
        int ep  = rem / 3;
        int ch  = rem - ep * 3;
        float e  = endpoints[(size_t)(blockBase + M) * 12 + rem];
        float s  = fast_sigmoid(e);
        float eq = s * 63.0f;
        // (eq*65536 + 32768)/64 * (31/64) == (eq*1024 + 512) * 0.484375
        ws[EU_OFF + (M * 3 + ch) * 4 + ep] = (eq * 1024.0f + 512.0f) * 0.484375f;
    }

    // ---- softmax over 32 logits for block m = lane&15 (2x redundant) -------
    const int m  = lane & 15;
    const int hi = lane >> 4;
    float f[32];
    {
        const float4* lp = (const float4*)(logits + (size_t)(blockBase + m) * 32);
        #pragma unroll
        for (int i = 0; i < 8; ++i) {
            float4 q = lp[i];
            f[4*i+0] = q.x; f[4*i+1] = q.y; f[4*i+2] = q.z; f[4*i+3] = q.w;
        }
    }
    float mx = f[0];
    #pragma unroll
    for (int k = 1; k < 32; ++k) mx = fmaxf(mx, f[k]);
    float sum = 0.0f;
    #pragma unroll
    for (int k = 0; k < 32; ++k) { f[k] = __expf(f[k] - mx); sum += f[k]; }
    const float inv = 1.0f / sum;

    // ---- pack A (16x32 f16): ISA 16-bit A layout ---------------------------
    // lane l: row M = l&15; half-slot j<8 -> K = hi*8+j ; j>=8 -> K = 16+hi*8+(j-8)
    v16h a;
    #pragma unroll
    for (int j = 0; j < 16; ++j) {
        int K = (j < 8) ? (hi * 8 + j) : (16 + hi * 8 + (j - 8));
        a[j] = (_Float16)(f[K] * inv);
    }
    // ---- pack B (32x16 f16): lane col n = l&15; j<8 -> K = hi*8+j ; j>=8 -> K = 16+hi*8+(j-8) ----
    v16h b;
    #pragma unroll
    for (int j = 0; j < 16; ++j)
        b[j] = (_Float16)bank[((j < 8) ? (hi * 8 + j) : (16 + hi * 8 + (j - 8))) * 16 + m];

    // ---- mask tile: D(16x16 f32) = A x B -----------------------------------
    v8f cacc = {};
    v8f dmat = __builtin_amdgcn_wmma_f32_16x16x32_f16(
        /*neg_a=*/false, a, /*neg_b=*/false, b,
        /*c_mod=*/(short)0, cacc, /*reuse_a=*/false, /*reuse_b=*/false);
    // dmat[r] = mask[block M = hi*8 + r][texel n = lane&15]

    // intra-wave LDS producer/consumer fence (no workgroup barrier needed)
    __builtin_amdgcn_wave_barrier();
    asm volatile("s_wait_dscnt 0" ::: "memory");
    __builtin_amdgcn_wave_barrier();

    // ---- per-texel decode: this lane owns texel n of 8 blocks --------------
    const int n = m;
    #pragma unroll
    for (int r = 0; r < 8; ++r) {
        const int M = hi * 8 + r;
        const float mask = dmat[r];

        // 3-bit index surrogate with LUT lerp
        float xi   = indices[(size_t)(blockBase + M) * 16 + n];
        float xs   = fast_sigmoid(xi) * 7.0f;
        float x0f  = floorf(xs);
        int   x0   = (int)x0f;
        int   x1   = x0 + 1; if (x1 > 7) x1 = 7;
        float frac = fminf(fmaxf(xs - x0f, 0.0f), 1.0f);
        float w0   = ws[LUT_OFF + x0];
        float w1   = ws[LUT_OFF + x1];
        float wgt  = w0 * (1.0f - frac) + w1 * frac;

        #pragma unroll
        for (int c = 0; c < 3; ++c) {
            float4 e4 = *(const float4*)&ws[EU_OFF + (M * 3 + c) * 4];
            float y1 = e4.x * (1.0f - wgt) + e4.y * wgt;   // subset 0
            float y2 = e4.z * (1.0f - wgt) + e4.w * wgt;   // subset 1
            float y  = mask * y1 + (1.0f - mask) * y2;
            // uf16 decode surrogate
            float hh = fminf(fmaxf(floorf((y - 1.0f) * (1.0f/1024.0f)) - 1.0f,
                                   0.0f), 31.0f);
            float v  = exp2f(hh - 14.0f) * (y * (1.0f/1024.0f) - hh);
            ws[Y_OFF + (c * 16 + M) * 18 + n] = v;
        }
    }

    __builtin_amdgcn_wave_barrier();
    asm volatile("s_wait_dscnt 0" ::: "memory");
    __builtin_amdgcn_wave_barrier();

    // ---- coalesced output: 12 segments x 256B per wave ---------------------
    // 16 consecutive blocks always share one block-row (16 | BLK_X).
    const int by  = blockBase >> 10;       // / BLK_X
    const int bx0 = blockBase & 1023;      // % BLK_X
    const int e   = lane * 2;              // element pair within 64-float segment
    const int Ms  = e >> 2;                // block within wave tile
    const int tc  = e & 3;                 // texel column (tc, tc+1 same block)
    #pragma unroll
    for (int c = 0; c < 3; ++c) {
        #pragma unroll
        for (int rr = 0; rr < 4; ++rr) {
            int t = rr * 4 + tc;
            float2 v2 = *(const float2*)&ws[Y_OFF + (c * 16 + Ms) * 18 + t];
            size_t o = (size_t)c * ((size_t)IMG_H * IMG_W)
                     + (size_t)(by * 4 + rr) * IMG_W
                     + (size_t)(bx0 * 4 + e);
            *(float2*)(out + o) = v2;
        }
    }
}

extern "C" void kernel_launch(void* const* d_in, const int* in_sizes, int n_in,
                              void* d_out, int out_size, void* d_ws, size_t ws_size,
                              hipStream_t stream) {
    const float* endpoints = (const float*)d_in[0];
    const float* indices   = (const float*)d_in[1];
    const float* logits    = (const float*)d_in[2];
    const float* bank      = (const float*)d_in[3];
    const float* lut       = (const float*)d_in[4];
    float* out = (float*)d_out;

    const int waves_total = NB / 16;                   // 65,536 waves
    const int wgs = waves_total / WAVES_PER_WG;        // 8,192 workgroups
    bc6_soft_decode<<<dim3(wgs), dim3(256), 0, stream>>>(
        endpoints, indices, logits, bank, lut, out);
}
